// CausalSelfAttention_40518721470731
// MI455X (gfx1250) — matrix operand order repaired and checked by reading, not compiled, not run on hardware
//
#include <hip/hip_runtime.h>
#include <hip/hip_bf16.h>
#include <math.h>

typedef __attribute__((ext_vector_type(16))) __bf16 v16bf;
typedef __attribute__((ext_vector_type(8)))  __bf16 v8bf;
typedef __attribute__((ext_vector_type(8)))  float  v8f;

#define WMMA_BF16(a, b, c) \
  __builtin_amdgcn_wmma_f32_16x16x32_bf16(false, (a), false, (b), (short)0, (c), false, false)

namespace cfg {
constexpr int B = 4, T = 2048, C = 1024, NH = 16, HD = 64;
constexpr int C3 = 3 * C;
}

__device__ __forceinline__ unsigned pack_bf16(float a, float b) {
  union { __bf16 h[2]; unsigned u; } u;
  u.h[0] = (__bf16)a;
  u.h[1] = (__bf16)b;
  return u.u;
}

// Issue one async HBM->LDS 16-byte copy (GVS mode: SGPR base + 32-bit lane
// offset). Tracked by ASYNCcnt; drain with async_wait() before the barrier.
__device__ __forceinline__ void async_copy_b128(unsigned lds_addr, unsigned voff,
                                                const void* sbase) {
  asm volatile("global_load_async_to_lds_b128 %0, %1, %2"
               :: "v"(lds_addr), "v"(voff), "s"(sbase)
               : "memory");
}
__device__ __forceinline__ void async_wait() {
  asm volatile("s_wait_asynccnt 0" ::: "memory");
}

// ---------------------------------------------------------------------------
// Fragment loaders for wave32 V_WMMA_F32_16X16X32_BF16 (CDNA5 ISA 7.12.2),
// each lowered to two ds_load_b128.
// A (16x32): lane (m, hf) = two contiguous 8-elem runs of row-major tile.
// B (32x16): lane (n, hf) = two contiguous 8-elem runs of transposed Bt[n][k].
// Row strides must be multiples of 8 elements for 16B alignment.
// ---------------------------------------------------------------------------
__device__ __forceinline__ v16bf load_a_frag(const __bf16* base, int ld) {
  const int lane = threadIdx.x & 31;
  const int m = lane & 15, hf = lane >> 4;
  union { v16bf v; v8bf h[2]; } f;
  const __bf16* p = base + m * ld + hf * 8;
  f.h[0] = *(const v8bf*)p;
  f.h[1] = *(const v8bf*)(p + 16);
  return f.v;
}

__device__ __forceinline__ v16bf load_bt_frag(const __bf16* baseT, int ld) {
  const int lane = threadIdx.x & 31;
  const int n = lane & 15, hf = lane >> 4;
  union { v16bf v; v8bf h[2]; } f;
  const __bf16* p = baseT + n * ld + hf * 8;
  f.h[0] = *(const v8bf*)p;
  f.h[1] = *(const v8bf*)(p + 16);
  return f.v;
}

// ---------------------------------------------------------------------------
// GEMM with bias: D[M,N] = A[M,K] * Bm[K,N] + bias[N].
// Block tile 128x128, BK=32; 8 waves = 4(M) x 2(N), each wave 32x64.
// All staging uses per-thread base pointers + constant offsets.
// ---------------------------------------------------------------------------
template <typename AT, typename OT>
__global__ void __launch_bounds__(256)
gemm_bias_wmma(const AT* __restrict__ A, const float* __restrict__ Bm,
               const float* __restrict__ bias, OT* __restrict__ D,
               int M, int N, int K) {
  constexpr int BM = 128, BN = 128, BK = 32;
  constexpr int LDA = BK + 8;   // 40 elems -> 16B-aligned runs
  constexpr int LDB = BK + 8;
  __shared__ __bf16 As[BM * LDA];
  __shared__ __bf16 Bt[BN * LDB];

  const int tid = threadIdx.x;
  const int wave = tid >> 5;
  const int wm = wave & 3;
  const int wn = wave >> 2;
  const int m0 = blockIdx.y * BM;
  const int n0 = blockIdx.x * BN;

  // Per-thread staging bases (hoisted out of the K loop).
  const int ra = tid >> 4, ja = tid & 15;       // A: row ra+16s, k-pair ja
  const AT* gA = A + (size_t)(m0 + ra) * K + 2 * ja;
  unsigned* asd = (unsigned*)As + ra * (LDA / 2) + ja;

  const int nb = tid & 127, jb = tid >> 7;      // B: col nb, k-pair jb+2s
  const float* gB = Bm + (size_t)(2 * jb) * N + n0 + nb;
  unsigned* btd = (unsigned*)Bt + nb * (LDB / 2) + jb;

  v8f acc[2][4];
#pragma unroll
  for (int i = 0; i < 2; ++i)
#pragma unroll
    for (int j = 0; j < 4; ++j)
#pragma unroll
      for (int e = 0; e < 8; ++e) acc[i][j][e] = 0.0f;

  for (int k0 = 0; k0 < K; k0 += BK) {
    // Stage A tile (128x32): 8 dwords per thread, constant offsets.
#pragma unroll
    for (int s = 0; s < 8; ++s) {
      unsigned pk;
      if constexpr (sizeof(AT) == 4) {
        const float2 f2 = *(const float2*)(gA + (size_t)s * 16 * K + k0);
        pk = pack_bf16(f2.x, f2.y);
      } else {
        pk = *(const unsigned*)(gA + (size_t)s * 16 * K + k0);
      }
      asd[s * 16 * (LDA / 2)] = pk;
    }
    // Stage B tile (32x128) transposed into Bt[n][k], packing k-pairs.
#pragma unroll
    for (int s = 0; s < 8; ++s) {
      const float b0 = gB[(size_t)(k0 + 4 * s) * N];
      const float b1 = gB[(size_t)(k0 + 4 * s) * N + N];
      btd[2 * s] = pack_bf16(b0, b1);
    }
    if (k0 + BK < K) {
      __builtin_prefetch(gA + k0 + BK, 0, 1);
      __builtin_prefetch(gB + (size_t)(k0 + BK) * N, 0, 1);
    }
    __syncthreads();

    v16bf af0 = load_a_frag(&As[(wm * 32 + 0) * LDA], LDA);
    v16bf af1 = load_a_frag(&As[(wm * 32 + 16) * LDA], LDA);
#pragma unroll
    for (int ni = 0; ni < 4; ++ni) {
      v16bf bf = load_bt_frag(&Bt[(wn * 64 + ni * 16) * LDB], LDB);
      acc[0][ni] = WMMA_BF16(af0, bf, acc[0][ni]);
      acc[1][ni] = WMMA_BF16(af1, bf, acc[1][ni]);
    }
    __syncthreads();
  }

  const int lane = tid & 31;
  const int n = lane & 15, hf = lane >> 4;
  OT* dp = D + (size_t)(m0 + wm * 32 + hf * 8) * N + n0 + wn * 64 + n;
  const float* bp = bias + n0 + wn * 64 + n;
#pragma unroll
  for (int mi = 0; mi < 2; ++mi)
#pragma unroll
    for (int ni = 0; ni < 4; ++ni) {
      const float bv = bp[ni * 16];
#pragma unroll
      for (int i = 0; i < 8; ++i)
        dp[(size_t)(mi * 16 + i) * N + ni * 16] = (OT)(acc[mi][ni][i] + bv);
    }
}

// ---------------------------------------------------------------------------
// Flash attention: one block per (64-query tile, head, batch); 4 waves, each
// owning a 16-query stripe. Q/K tiles staged HBM->LDS with
// global_load_async_to_lds_b128; V transposed manually for the B^T layout.
// qkv ws: [B*T, 3C] bf16;  y ws: [B*T, C] bf16 (col = h*HD + d).
// ---------------------------------------------------------------------------
__global__ void __launch_bounds__(128)
flash_attn_wmma(const __bf16* __restrict__ qkv, __bf16* __restrict__ y) {
  using namespace cfg;
  constexpr int BQ = 64, BKV = 64;
  constexpr int LQ = HD + 8;    // 72 (row = 144 B, 16B-aligned chunks)
  constexpr int LK = HD + 8;    // Ks[key][d]
  constexpr int LV = BKV + 8;   // Vt[d][key]
  constexpr int LP = BKV + 8;   // Ps[q][key]
  constexpr int LS = BKV + 4;   // Ss[q][key] (f32)
  __shared__ __bf16 Qs[BQ * LQ];
  __shared__ __bf16 Ks[BKV * LK];
  __shared__ __bf16 Vt[HD * LV];
  __shared__ __bf16 Ps[BQ * LP];
  __shared__ float  Ss[BQ * LS];
  __shared__ float  m_s[BQ], l_s[BQ], a_s[BQ];

  const int qt = blockIdx.x, h = blockIdx.y, b = blockIdx.z;
  const int tid = threadIdx.x, wave = tid >> 5, lane = tid & 31;
  const int n = lane & 15, hf = lane >> 4;
  const size_t bt0 = (size_t)b * T;

  // --- async staging geometry: 16B chunks; row = 8 chunks of 64 bf16 ------
  const int cp = tid & 7;            // chunk within row
  const int row0 = tid >> 3;         // rows row0 + 16s, s = 0..3
  const unsigned voff0 = (unsigned)(row0 * (C3 * 2) + cp * 16);
  const unsigned ldsQ0 = (unsigned)(size_t)&Qs[row0 * LQ + cp * 8];
  const unsigned ldsK0 = (unsigned)(size_t)&Ks[row0 * LK + cp * 8];

  // Q tile: async copy, base = row (b, qt*BQ), col h*HD.
  {
    const __bf16* q_base = qkv + (bt0 + (size_t)qt * BQ) * C3 + h * HD;
#pragma unroll
    for (int s = 0; s < 4; ++s)
      async_copy_b128(ldsQ0 + (unsigned)(s * 16 * LQ * 2),
                      voff0 + (unsigned)(s * 16 * C3 * 2), q_base);
  }
  if (tid < BQ) { m_s[tid] = -INFINITY; l_s[tid] = 0.0f; }

  // V staging bases: dword dpv of row keyv0+4s (transposed into Vt[d][key]).
  const int dpv = tid & 31;
  const int keyv0 = tid >> 5;
  const unsigned* gV0 = (const unsigned*)qkv +
      (bt0 + keyv0) * (size_t)(C3 / 2) + C + h * (HD / 2) + dpv;

  v8f o[4];
#pragma unroll
  for (int j = 0; j < 4; ++j)
#pragma unroll
    for (int e = 0; e < 8; ++e) o[j][e] = 0.0f;

  const float scale = 0.125f;  // 1/sqrt(HD)

  for (int kt = 0; kt <= qt; ++kt) {
    // K tile: async HBM->LDS, natural [key][d] order (= B^T for S).
    const __bf16* k_base = qkv + (bt0 + (size_t)kt * BKV) * C3 + C + h * HD;
#pragma unroll
    for (int s = 0; s < 4; ++s)
      async_copy_b128(ldsK0 + (unsigned)(s * 16 * LK * 2),
                      voff0 + (unsigned)(s * 16 * C3 * 2), k_base);

    // V tile: manual transpose into Vt[d][key] (16 dwords per thread).
    {
      const unsigned* gV = gV0 + (size_t)kt * BKV * (C3 / 2);
      __bf16* vd = &Vt[(2 * dpv) * LV + keyv0];
#pragma unroll
      for (int s = 0; s < 16; ++s) {
        union { unsigned u; __bf16 hh[2]; } w;
        w.u = gV[(size_t)s * 4 * (C3 / 2)];
        vd[s * 4]      = w.hh[0];
        vd[s * 4 + LV] = w.hh[1];
      }
    }
    async_wait();
    __syncthreads();

    // S stripe: 16 query rows x 64 keys; K-dim = HD (2 WMMA steps).
    v16bf qa0 = load_a_frag(&Qs[(wave * 16) * LQ], LQ);
    v16bf qa1 = load_a_frag(&Qs[(wave * 16) * LQ + 32], LQ);
#pragma unroll
    for (int ni = 0; ni < 4; ++ni) {
      v8f s;
#pragma unroll
      for (int e = 0; e < 8; ++e) s[e] = 0.0f;
      v16bf kb0 = load_bt_frag(&Ks[(ni * 16) * LK], LK);
      v16bf kb1 = load_bt_frag(&Ks[(ni * 16) * LK + 32], LK);
      s = WMMA_BF16(qa0, kb0, s);
      s = WMMA_BF16(qa1, kb1, s);
#pragma unroll
      for (int i = 0; i < 8; ++i)
        Ss[(wave * 16 + hf * 8 + i) * LS + ni * 16 + n] = s[i] * scale;
    }
    __syncthreads();

    // Online softmax: one thread per query row; causal mask on diagonal tile.
    if (tid < BQ) {
      const int r = tid;
      const int kmax = (kt == qt) ? (r + 1) : BKV;
      const float* srow = &Ss[r * LS];
      unsigned* prow = (unsigned*)Ps + r * (LP / 2);
      const float mold = m_s[r];
      float mt = -INFINITY;
#pragma unroll
      for (int j = 0; j < BKV; ++j) {
        const float v = srow[j];
        mt = (j < kmax) ? fmaxf(mt, v) : mt;
      }
      const float mnew = fmaxf(mold, mt);
      const float alpha = __expf(mold - mnew);
      float sum = 0.0f;
#pragma unroll
      for (int j = 0; j < BKV / 2; ++j) {
        const float p0 = (2 * j < kmax) ? __expf(srow[2 * j] - mnew) : 0.0f;
        const float p1 = (2 * j + 1 < kmax) ? __expf(srow[2 * j + 1] - mnew) : 0.0f;
        sum += p0 + p1;
        prow[j] = pack_bf16(p0, p1);
      }
      l_s[r] = l_s[r] * alpha + sum;
      m_s[r] = mnew;
      a_s[r] = alpha;
    }
    __syncthreads();

    // O = alpha*O + P V  (K-dim = 64 keys, 2 WMMA steps).
    v16bf pa0 = load_a_frag(&Ps[(wave * 16) * LP], LP);
    v16bf pa1 = load_a_frag(&Ps[(wave * 16) * LP + 32], LP);
    float alf[8];
#pragma unroll
    for (int i = 0; i < 8; ++i) alf[i] = a_s[wave * 16 + hf * 8 + i];
#pragma unroll
    for (int ni = 0; ni < 4; ++ni) {
      v16bf vb0 = load_bt_frag(&Vt[(ni * 16) * LV], LV);
      v16bf vb1 = load_bt_frag(&Vt[(ni * 16) * LV + 32], LV);
#pragma unroll
      for (int i = 0; i < 8; ++i) o[ni][i] *= alf[i];
      o[ni] = WMMA_BF16(pa0, vb0, o[ni]);
      o[ni] = WMMA_BF16(pa1, vb1, o[ni]);
    }
    __syncthreads();
  }

  // Normalize and write y (bf16 workspace), base pointer + folded offsets.
  __bf16* yp = y + (bt0 + (size_t)qt * BQ + wave * 16 + hf * 8) * C + h * HD + n;
  const float* lp = &l_s[wave * 16 + hf * 8];
#pragma unroll
  for (int ni = 0; ni < 4; ++ni)
#pragma unroll
    for (int i = 0; i < 8; ++i)
      yp[(size_t)i * C + ni * 16] = (__bf16)(o[ni][i] / lp[i]);
}

// ---------------------------------------------------------------------------
extern "C" void kernel_launch(void* const* d_in, const int* in_sizes, int n_in,
                              void* d_out, int out_size, void* d_ws, size_t ws_size,
                              hipStream_t stream) {
  using namespace cfg;
  (void)in_sizes; (void)n_in; (void)out_size; (void)ws_size;
  const float* x      = (const float*)d_in[0];
  const float* w_qkv  = (const float*)d_in[1];
  const float* b_qkv  = (const float*)d_in[2];
  const float* w_proj = (const float*)d_in[3];
  const float* b_proj = (const float*)d_in[4];
  float* out = (float*)d_out;

  __bf16* qkv_ws = (__bf16*)d_ws;                // [B*T, 3C] bf16 (48 MB)
  __bf16* y_ws   = qkv_ws + (size_t)B * T * C3;  // [B*T, C]  bf16 (16 MB)

  const int M = B * T;  // 8192

  // 1) qkv = x @ w_qkv + b_qkv   (fp32 in -> bf16 ws)
  gemm_bias_wmma<float, __bf16>
      <<<dim3(C3 / 128, M / 128), 256, 0, stream>>>(x, w_qkv, b_qkv, qkv_ws, M, C3, C);

  // 2) causal flash attention per (q-tile, head, batch)
  flash_attn_wmma<<<dim3(T / 64, NH, B), 128, 0, stream>>>(qkv_ws, y_ws);

  // 3) out = y @ w_proj + b_proj (bf16 ws -> fp32 out)
  gemm_bias_wmma<__bf16, float>
      <<<dim3(C / 128, M / 128), 256, 0, stream>>>(y_ws, w_proj, b_proj, out, M, C, C);
}
